// PagedDecoderLayer_83880711291164
// MI455X (gfx1250) — compile-verified, matrix operand order checked by reading, not run on hardware
//
#include <hip/hip_runtime.h>

// ---------------------------------------------------------------------------
// Types / WMMA helpers (CDNA5 gfx1250, wave32)
// ---------------------------------------------------------------------------
typedef __attribute__((ext_vector_type(16))) __bf16 bf16x16;
typedef __attribute__((ext_vector_type(8)))  float  f32x8;
typedef __attribute__((ext_vector_type(4)))  float  f32x4;
typedef int v4i_gcc __attribute__((vector_size(16)));

union FragAB { bf16x16 v; f32x4 q[2]; };   // 16x32 (A) or 32x16 (B) bf16 fragment, 8 VGPRs
union FragC  { f32x8  v; float f[8];  };   // 16x16 f32 C/D fragment, 8 VGPRs

#define WMMA_BF16(A, B, C) \
  __builtin_amdgcn_wmma_f32_16x16x32_bf16(false, (A), false, (B), (short)0, (C), false, false)

// Model dims
#define TSEQ 2048
#define HID  2048
#define NH   16
#define NHK  8
#define DH   128
#define FF   6144

// ---------------------------------------------------------------------------
// gfx1250 async global->LDS copy (ASYNCcnt-tracked DMA), with sync fallback
// Builtin signature (from compiler diagnostic): (v4i* src, v4i as3* dst, imm, imm)
// ---------------------------------------------------------------------------
__device__ __forceinline__ void async_copy16(const void* g, void* l) {
#if defined(__gfx1250__) && __has_builtin(__builtin_amdgcn_global_load_async_to_lds_b128)
  __builtin_amdgcn_global_load_async_to_lds_b128(
      (v4i_gcc*)g,
      (__attribute__((address_space(3))) v4i_gcc*)l, 0, 0);
#else
  *(f32x4*)l = *(const f32x4*)g;
#endif
}

__device__ __forceinline__ void async_wait0() {
#if defined(__gfx1250__) && __has_builtin(__builtin_amdgcn_s_wait_asynccnt)
  __builtin_amdgcn_s_wait_asynccnt(0);
#elif defined(__gfx1250__) && __has_builtin(__builtin_amdgcn_global_load_async_to_lds_b128)
  asm volatile("s_wait_asynccnt 0" ::: "memory");
#endif
}

// ---------------------------------------------------------------------------
// rmsnorm(x, w) -> bf16, one block per row
// ---------------------------------------------------------------------------
__global__ __launch_bounds__(256) void rmsnorm_to_bf16(
    const float* __restrict__ x, const float* __restrict__ w,
    __bf16* __restrict__ out, int ncols)
{
  const int row = blockIdx.x;
  const int tid = threadIdx.x, lane = tid & 31, wid = tid >> 5;
  const float* xr = x + (size_t)row * ncols;
  float ss = 0.f;
  for (int i = tid; i < ncols; i += 256) { float v = xr[i]; ss += v * v; }
#pragma unroll
  for (int off = 16; off; off >>= 1) ss += __shfl_xor(ss, off, 32);
  __shared__ float red[8];
  if (lane == 0) red[wid] = ss;
  __syncthreads();
  float tot = 0.f;
#pragma unroll
  for (int k = 0; k < 8; ++k) tot += red[k];
  const float rinv = rsqrtf(tot / (float)ncols + 1e-6f);
  __bf16* orow = out + (size_t)row * ncols;
  for (int i = tid; i < ncols; i += 256) orow[i] = (__bf16)(xr[i] * rinv * w[i]);
}

// ---------------------------------------------------------------------------
// Per-(t, head) rmsnorm + RoPE -> bf16. One wave per head, one block per t.
// ---------------------------------------------------------------------------
__global__ __launch_bounds__(256) void qknorm_rope_bf16(
    const float* __restrict__ x, const float* __restrict__ nw,
    const int* __restrict__ positions, __bf16* __restrict__ out, int nheads)
{
  const int t = blockIdx.x;
  const int lane = threadIdx.x & 31, wid = threadIdx.x >> 5;
  const float pos = (float)positions[t];
  for (int h = wid; h < nheads; h += 8) {
    const float* xr = x + ((size_t)t * nheads + h) * DH;
    float a0 = xr[lane * 2], a1 = xr[lane * 2 + 1];
    float b0 = xr[64 + lane * 2], b1 = xr[64 + lane * 2 + 1];
    float ss = a0 * a0 + a1 * a1 + b0 * b0 + b1 * b1;
#pragma unroll
    for (int off = 16; off; off >>= 1) ss += __shfl_xor(ss, off, 32);
    const float rinv = rsqrtf(ss * (1.f / 128.f) + 1e-6f);
    __bf16* orow = out + ((size_t)t * nheads + h) * DH;
    float a[2] = {a0, a1}, b[2] = {b0, b1};
#pragma unroll
    for (int j = 0; j < 2; ++j) {
      int d = lane * 2 + j;
      float an = a[j] * rinv * nw[d];
      float bn = b[j] * rinv * nw[64 + d];
      // inv = THETA^(-d/64); THETA = 1e6 -> log2(1e6) = 19.931568569324174
      float fr = pos * exp2f(-(float)d * (19.9315685693f / 64.f));
      float c = cosf(fr), s = sinf(fr);
      orow[d]      = (__bf16)(an * c - bn * s);
      orow[64 + d] = (__bf16)(bn * c + an * s);
    }
  }
}

// ---------------------------------------------------------------------------
// B-fragment from an f32 LDS row (32 K values), converted to bf16 in-register.
// ---------------------------------------------------------------------------
__device__ __forceinline__ FragAB cvt_bfrag_f32(const float* base, int hlf) {
  f32x4 a0 = *(const f32x4*)(base + hlf * 8);
  f32x4 a1 = *(const f32x4*)(base + hlf * 8 + 4);
  f32x4 b0 = *(const f32x4*)(base + 16 + hlf * 8);
  f32x4 b1 = *(const f32x4*)(base + 16 + hlf * 8 + 4);
  union { bf16x16 v; __bf16 t[16]; } u;
  u.t[0]  = (__bf16)a0.x; u.t[1]  = (__bf16)a0.y; u.t[2]  = (__bf16)a0.z; u.t[3]  = (__bf16)a0.w;
  u.t[4]  = (__bf16)a1.x; u.t[5]  = (__bf16)a1.y; u.t[6]  = (__bf16)a1.z; u.t[7]  = (__bf16)a1.w;
  u.t[8]  = (__bf16)b0.x; u.t[9]  = (__bf16)b0.y; u.t[10] = (__bf16)b0.z; u.t[11] = (__bf16)b0.w;
  u.t[12] = (__bf16)b1.x; u.t[13] = (__bf16)b1.y; u.t[14] = (__bf16)b1.z; u.t[15] = (__bf16)b1.w;
  FragAB f; f.v = u.v; return f;
}

// ---------------------------------------------------------------------------
// Tiled NT GEMM: C[M,N] = A(bf16, MxK row-major) * W(f32, NxK row-major)^T
// 128x128 block tile, BK=32, 256 threads (8 waves, 2x4), WMMA bf16.
// Double-buffered async global->LDS staging; W converted at fragment load.
// ---------------------------------------------------------------------------
enum { EPI_F32 = 0, EPI_BF16 = 1, EPI_SILU_BF16 = 2, EPI_MUL_BF16 = 3, EPI_ADD_F32 = 4 };

template <int EPI>
__global__ __launch_bounds__(256) void gemm_nt_bf16(
    const __bf16* __restrict__ A, const float* __restrict__ W,
    void* __restrict__ outp, const void* __restrict__ aux,
    int M, int N, int K)
{
  constexpr int BM = 128, BN = 128, BK = 32;
  constexpr int BKP = 40;  // bf16 padded row (80B, conflict-free fragment reads)
  constexpr int WP  = 36;  // f32 padded row (144B)
  __shared__ __attribute__((aligned(16))) __bf16 As[2][BM * BKP];  // 2 x 10 KB
  __shared__ __attribute__((aligned(16))) float  Ws[2][BN * WP];   // 2 x 18 KB
  const int tid = threadIdx.x;
  const int lane = tid & 31, wid = tid >> 5;
  const int hlf = lane >> 4, l16 = lane & 15;
  const int m0 = blockIdx.y * BM, n0 = blockIdx.x * BN;
  const int wm = (wid >> 2) * 64, wn = (wid & 3) * 32;

  FragC acc[4][2];
#pragma unroll
  for (int i = 0; i < 4; ++i)
#pragma unroll
    for (int j = 0; j < 2; ++j)
#pragma unroll
      for (int e = 0; e < 8; ++e) acc[i][j].f[e] = 0.f;

  auto stage = [&](int k0, int buf) {
    // A tile: 128x32 bf16 = 512 x 16B chunks
#pragma unroll
    for (int i = 0; i < 2; ++i) {
      int c = tid + 256 * i;
      int r = c >> 2, kc = (c & 3) * 8;
      async_copy16(A + (size_t)(m0 + r) * K + k0 + kc, &As[buf][r * BKP + kc]);
    }
    // W tile: 128x32 f32 = 1024 x 16B chunks (raw copy, converted on read)
#pragma unroll
    for (int i = 0; i < 4; ++i) {
      int c = tid + 256 * i;
      int r = c >> 3, kc = (c & 7) * 4;
      async_copy16(W + (size_t)(n0 + r) * K + k0 + kc, &Ws[buf][r * WP + kc]);
    }
  };

  stage(0, 0);
  async_wait0();
  __syncthreads();
  int cur = 0;

  for (int k0 = 0; k0 < K; k0 += BK) {
    if (k0 + BK < K) stage(k0 + BK, cur ^ 1);  // overlap DMA with compute

    FragAB af[4], wf[2];
#pragma unroll
    for (int i = 0; i < 4; ++i) {
      int row = wm + i * 16 + l16;
      af[i].q[0] = *(const f32x4*)&As[cur][row * BKP + hlf * 8];
      af[i].q[1] = *(const f32x4*)&As[cur][row * BKP + 16 + hlf * 8];
    }
#pragma unroll
    for (int j = 0; j < 2; ++j) {
      int row = wn + j * 16 + l16;
      wf[j] = cvt_bfrag_f32(&Ws[cur][row * WP], hlf);
    }
#pragma unroll
    for (int i = 0; i < 4; ++i)
#pragma unroll
      for (int j = 0; j < 2; ++j)
        acc[i][j].v = WMMA_BF16(af[i].v, wf[j].v, acc[i][j].v);

    async_wait0();
    __syncthreads();
    cur ^= 1;
  }

  // Epilogue: C frag element (r, lane): m = r + 8*(lane/16), n = lane%16
#pragma unroll
  for (int i = 0; i < 4; ++i)
#pragma unroll
    for (int j = 0; j < 2; ++j)
#pragma unroll
      for (int r = 0; r < 8; ++r) {
        int gm = m0 + wm + i * 16 + r + 8 * hlf;
        int gn = n0 + wn + j * 16 + l16;
        size_t idx = (size_t)gm * N + gn;
        float c = acc[i][j].f[r];
        if constexpr (EPI == EPI_F32) {
          ((float*)outp)[idx] = c;
        } else if constexpr (EPI == EPI_BF16) {
          ((__bf16*)outp)[idx] = (__bf16)c;
        } else if constexpr (EPI == EPI_SILU_BF16) {
          ((__bf16*)outp)[idx] = (__bf16)(c / (1.f + __expf(-c)));
        } else if constexpr (EPI == EPI_MUL_BF16) {
          float g = (float)((const __bf16*)aux)[idx];
          ((__bf16*)outp)[idx] = (__bf16)(c * g);
        } else {  // EPI_ADD_F32
          ((float*)outp)[idx] = c + ((const float*)aux)[idx];
        }
      }
}

// ---------------------------------------------------------------------------
// Flash attention, causal GQA (2 Q heads per KV head), bf16 WMMA, f32 softmax.
// Block = 128 threads (4 waves), 64 query rows per block, 64-key blocks.
// Q/K tiles staged with async DMA; fragments preloaded in bursts.
// ---------------------------------------------------------------------------
__global__ __launch_bounds__(128) void flash_attn_bf16(
    const __bf16* __restrict__ Qg, const __bf16* __restrict__ Kg,
    const __bf16* __restrict__ Vg, __bf16* __restrict__ Og)
{
  constexpr int BQ = 64, BS = 64;
  constexpr int QP = 136, KP = 136, VP = 72, PP = 72;  // padded LDS strides
  constexpr float SCALE = 0.08838834764831845f;        // D^-0.5
  __shared__ __attribute__((aligned(16))) __bf16 Qs[BQ * QP];
  __shared__ __attribute__((aligned(16))) __bf16 Ks[BS * KP];
  __shared__ __attribute__((aligned(16))) __bf16 Vts[DH * VP];     // V transposed [d][s]
  __shared__ __attribute__((aligned(16))) __bf16 Ps[4 * 16 * PP];  // per-wave P tile

  const int t0 = blockIdx.x * BQ;
  const int hq = blockIdx.y;
  const int hk = hq >> 1;
  const int tid = threadIdx.x, lane = tid & 31, wid = tid >> 5;
  const int hlf = lane >> 4, l16 = lane & 15;
  const __bf16* qp = Qg + (size_t)hq * DH;
  const __bf16* kp = Kg + (size_t)hk * DH;
  const __bf16* vp = Vg + (size_t)hk * DH;

  // Stage Q tile (64 x 128) via async DMA
#pragma unroll
  for (int i = 0; i < 8; ++i) {
    int c = tid + 128 * i;
    int r = c >> 4, dc = (c & 15) * 8;
    async_copy16(qp + (size_t)(t0 + r) * (NH * DH) + dc, &Qs[r * QP + dc]);
  }
  async_wait0();
  __syncthreads();

  FragAB qf[4];
#pragma unroll
  for (int d = 0; d < 4; ++d) {
    int row = wid * 16 + l16;
    qf[d].q[0] = *(const f32x4*)&Qs[row * QP + d * 32 + hlf * 8];
    qf[d].q[1] = *(const f32x4*)&Qs[row * QP + d * 32 + 16 + hlf * 8];
  }

  FragC o[8];
#pragma unroll
  for (int n = 0; n < 8; ++n)
#pragma unroll
    for (int e = 0; e < 8; ++e) o[n].f[e] = 0.f;
  float mrow[8], lrow[8];
#pragma unroll
  for (int r = 0; r < 8; ++r) { mrow[r] = -3.0e38f; lrow[r] = 0.f; }

  for (int s0 = 0; s0 <= t0; s0 += BS) {
    __syncthreads();  // protect K/Vt from previous iteration's readers
    // Stage K (64x128, async) and V^T (128x64, transposing scatter)
#pragma unroll
    for (int i = 0; i < 8; ++i) {
      int c = tid + 128 * i;
      int r = c >> 4, dc = (c & 15) * 8;
      async_copy16(kp + (size_t)(s0 + r) * (NHK * DH) + dc, &Ks[r * KP + dc]);
      f32x4 vv = *(const f32x4*)(vp + (size_t)(s0 + r) * (NHK * DH) + dc);
      __bf16 tmp[8];
      *(f32x4*)tmp = vv;
#pragma unroll
      for (int e = 0; e < 8; ++e) Vts[(dc + e) * VP + r] = tmp[e];
    }
    async_wait0();
    __syncthreads();

    // S = Q K^T (16 query rows per wave x 64 keys); preload 4 K-frags per d
    FragC sf[4];
#pragma unroll
    for (int j = 0; j < 4; ++j)
#pragma unroll
      for (int e = 0; e < 8; ++e) sf[j].f[e] = 0.f;
#pragma unroll
    for (int d = 0; d < 4; ++d) {
      FragAB kf[4];
#pragma unroll
      for (int j = 0; j < 4; ++j) {
        int row = j * 16 + l16;
        kf[j].q[0] = *(const f32x4*)&Ks[row * KP + d * 32 + hlf * 8];
        kf[j].q[1] = *(const f32x4*)&Ks[row * KP + d * 32 + 16 + hlf * 8];
      }
#pragma unroll
      for (int j = 0; j < 4; ++j)
        sf[j].v = WMMA_BF16(qf[d].v, kf[j].v, sf[j].v);
    }

    // Scale + causal mask (only diagonal block can mask)
#pragma unroll
    for (int j = 0; j < 4; ++j)
#pragma unroll
      for (int r = 0; r < 8; ++r) {
        float v = sf[j].f[r] * SCALE;
        if (s0 == t0) {
          int sg = s0 + j * 16 + l16;
          int tg = t0 + wid * 16 + r + 8 * hlf;
          if (sg > tg) v = -1.0e30f;
        }
        sf[j].f[r] = v;
      }

    // Online softmax (rows r + 8*hlf; reduce across the 16-lane half)
#pragma unroll
    for (int r = 0; r < 8; ++r) {
      float rmax = fmaxf(fmaxf(sf[0].f[r], sf[1].f[r]), fmaxf(sf[2].f[r], sf[3].f[r]));
#pragma unroll
      for (int off = 8; off; off >>= 1) rmax = fmaxf(rmax, __shfl_xor(rmax, off, 32));
      float mn = fmaxf(mrow[r], rmax);
      float corr = __expf(mrow[r] - mn);
      float rs = 0.f;
#pragma unroll
      for (int j = 0; j < 4; ++j) {
        float p = __expf(sf[j].f[r] - mn);
        sf[j].f[r] = p;
        rs += p;
      }
#pragma unroll
      for (int off = 8; off; off >>= 1) rs += __shfl_xor(rs, off, 32);
      lrow[r] = lrow[r] * corr + rs;
      mrow[r] = mn;
#pragma unroll
      for (int n = 0; n < 8; ++n) o[n].f[r] *= corr;
    }

    // P -> LDS (bf16), then O += P V via WMMA (preload V-frags in groups of 4)
    __bf16* pw = &Ps[wid * 16 * PP];
#pragma unroll
    for (int j = 0; j < 4; ++j)
#pragma unroll
      for (int r = 0; r < 8; ++r)
        pw[(r + 8 * hlf) * PP + j * 16 + l16] = (__bf16)sf[j].f[r];
    __syncthreads();  // order bf16 stores vs vector loads below
#pragma unroll
    for (int kk = 0; kk < 2; ++kk) {
      FragAB pa;
      pa.q[0] = *(const f32x4*)&pw[l16 * PP + kk * 32 + hlf * 8];
      pa.q[1] = *(const f32x4*)&pw[l16 * PP + kk * 32 + 16 + hlf * 8];
#pragma unroll
      for (int g = 0; g < 2; ++g) {
        FragAB vf[4];
#pragma unroll
        for (int j = 0; j < 4; ++j) {
          int row = (g * 4 + j) * 16 + l16;
          vf[j].q[0] = *(const f32x4*)&Vts[row * VP + kk * 32 + hlf * 8];
          vf[j].q[1] = *(const f32x4*)&Vts[row * VP + kk * 32 + 16 + hlf * 8];
        }
#pragma unroll
        for (int j = 0; j < 4; ++j)
          o[g * 4 + j].v = WMMA_BF16(pa.v, vf[j].v, o[g * 4 + j].v);
      }
    }
  }

  // Epilogue: O /= l, store bf16 [t][hq*128 + d]
#pragma unroll
  for (int r = 0; r < 8; ++r) {
    float linv = 1.f / lrow[r];
    int t = t0 + wid * 16 + r + 8 * hlf;
#pragma unroll
    for (int n = 0; n < 8; ++n) {
      int d = n * 16 + l16;
      Og[(size_t)t * (NH * DH) + hq * DH + d] = (__bf16)(o[n].f[r] * linv);
    }
  }
}

// ---------------------------------------------------------------------------
// Host launcher
// ---------------------------------------------------------------------------
extern "C" void kernel_launch(void* const* d_in, const int* in_sizes, int n_in,
                              void* d_out, int out_size, void* d_ws, size_t ws_size,
                              hipStream_t stream)
{
  (void)in_sizes; (void)n_in; (void)out_size; (void)ws_size;
  const int*   positions = (const int*)  d_in[0];
  const float* hidden    = (const float*)d_in[1];
  const float* wq  = (const float*)d_in[2];
  const float* wk  = (const float*)d_in[3];
  const float* wv  = (const float*)d_in[4];
  const float* wo  = (const float*)d_in[5];
  const float* qnw = (const float*)d_in[6];
  const float* knw = (const float*)d_in[7];
  const float* ln1 = (const float*)d_in[8];
  const float* ln2 = (const float*)d_in[9];
  const float* wg  = (const float*)d_in[10];
  const float* wu  = (const float*)d_in[11];
  const float* wd  = (const float*)d_in[12];
  float* out = (float*)d_out;

  char* ws = (char*)d_ws;
  __bf16* hn   = (__bf16*)(ws + 0);          // T*HID bf16        (8 MB)
  float*  qf32 = (float*)(ws + 8388608);     // T*H*D f32        (16 MB)
  float*  kf32 = (float*)(ws + 25165824);    // T*HK*D f32        (8 MB)
  __bf16* qb   = (__bf16*)(ws + 33554432);   // T*H*D bf16        (8 MB)
  __bf16* kb   = (__bf16*)(ws + 41943040);   // T*HK*D bf16       (4 MB)
  __bf16* vb   = (__bf16*)(ws + 46137344);   // T*HK*D bf16       (4 MB)
  __bf16* ab   = (__bf16*)(ws + 50331648);   // T*H*D bf16 attn   (8 MB)
  __bf16* h2n  = (__bf16*)(ws + 58720256);   // T*HID bf16        (8 MB)
  __bf16* gb   = (__bf16*)(ws + 67108864);   // T*FF bf16        (24 MB)
  __bf16* actb = (__bf16*)(ws + 92274688);   // T*FF bf16        (24 MB)

  // 1) ln1
  rmsnorm_to_bf16<<<TSEQ, 256, 0, stream>>>(hidden, ln1, hn, HID);
  // 2) QKV projections (v goes straight to bf16)
  gemm_nt_bf16<EPI_F32 ><<<dim3(16, 16), 256, 0, stream>>>(hn, wq, qf32, nullptr, TSEQ, NH * DH, HID);
  gemm_nt_bf16<EPI_F32 ><<<dim3( 8, 16), 256, 0, stream>>>(hn, wk, kf32, nullptr, TSEQ, NHK * DH, HID);
  gemm_nt_bf16<EPI_BF16><<<dim3( 8, 16), 256, 0, stream>>>(hn, wv, vb,   nullptr, TSEQ, NHK * DH, HID);
  // 3) qk-norm + RoPE -> bf16
  qknorm_rope_bf16<<<TSEQ, 256, 0, stream>>>(qf32, qnw, positions, qb, NH);
  qknorm_rope_bf16<<<TSEQ, 256, 0, stream>>>(kf32, knw, positions, kb, NHK);
  // 4) causal flash attention
  flash_attn_bf16<<<dim3(TSEQ / 64, NH), 128, 0, stream>>>(qb, kb, vb, ab);
  // 5) out-proj + residual (-> d_out)
  gemm_nt_bf16<EPI_ADD_F32><<<dim3(16, 16), 256, 0, stream>>>(ab, wo, out, hidden, TSEQ, HID, NH * DH);
  // 6) ln2
  rmsnorm_to_bf16<<<TSEQ, 256, 0, stream>>>(out, ln2, h2n, HID);
  // 7) SwiGLU MLP: silu(gate), *up, down + residual
  gemm_nt_bf16<EPI_SILU_BF16><<<dim3(48, 16), 256, 0, stream>>>(h2n, wg, gb,   nullptr, TSEQ, FF, HID);
  gemm_nt_bf16<EPI_MUL_BF16 ><<<dim3(48, 16), 256, 0, stream>>>(h2n, wu, actb, gb,      TSEQ, FF, HID);
  gemm_nt_bf16<EPI_ADD_F32  ><<<dim3(16, 16), 256, 0, stream>>>(actb, wd, out, out,     TSEQ, HID, FF);
}
